// CovariateAttention_67516885893229
// MI455X (gfx1250) — compile-verified, matrix operand order checked by reading, not run on hardware
//
#include <hip/hip_runtime.h>
#include <hip/hip_bf16.h>
#include <math.h>

typedef __attribute__((ext_vector_type(16))) __bf16 v16bf;
typedef __attribute__((ext_vector_type(8)))  float  v8f;
typedef __attribute__((ext_vector_type(4)))  unsigned int u32x4;
typedef __attribute__((ext_vector_type(8)))  int i32x8;
typedef __attribute__((ext_vector_type(4)))  int i32x4;

static constexpr int S_  = 2048;
static constexpr int B_  = 2;
static constexpr int E_  = 1024;   // embed == d_attn
static constexpr int H_  = 16;
static constexpr int HD_ = 64;
static constexpr int M_  = B_ * S_; // 4096 rows

#if __has_builtin(__builtin_amdgcn_tensor_load_to_lds) && \
    __has_builtin(__builtin_amdgcn_s_wait_tensorcnt)
#define USE_TDM 1
#endif

__device__ __forceinline__ v8f wmma_bf16(v16bf a, v16bf b, v8f c) {
  // (neg_a, A, neg_b, B, c_mod, C, reuse_a, reuse_b)
  return __builtin_amdgcn_wmma_f32_16x16x32_bf16(false, a, false, b,
                                                 (short)0, c, false, false);
}

// ---------------------------------------------------------------------------
// TDM: async 2D tile load global -> LDS (bf16 elements, row-major dest).
// D# packing per CDNA5 ISA §8.3 (group0) / §8.4 (group1).
// tile0 = contiguous (x) extent, tile1 = rows; stride0 in elements.
// ---------------------------------------------------------------------------
__device__ __forceinline__ void tdm_load_2d_bf16(
    const void* gptr, unsigned ldsOff,
    unsigned tile0, unsigned tile1,
    unsigned tdim0, unsigned tdim1, unsigned stride0)
{
#ifdef USE_TDM
  u32x4 g0 = {};
  g0.x = 1u;                                   // count=1, user descriptor
  g0.y = ldsOff;                               // lds_addr (bytes)
  unsigned long long ga = (unsigned long long)(uintptr_t)gptr;
  g0.z = (unsigned)ga;                         // global_addr[31:0]
  g0.w = (unsigned)((ga >> 32) & 0x01FFFFFFull) | 0x80000000u; // addr[56:32] | type=2

  i32x8 g1 = {};
  g1[0] = (int)(1u << 16);                     // wg_mask=0, data_size=1 (2 bytes)
  g1[1] = (int)((tdim0 & 0xFFFFu) << 16);      // tensor_dim0[15:0] @ bits 63:48
  g1[2] = (int)(((tdim0 >> 16) & 0xFFFFu) |    // tensor_dim0[31:16]
                ((tdim1 & 0xFFFFu) << 16));    // tensor_dim1[15:0]
  g1[3] = (int)(((tdim1 >> 16) & 0xFFFFu) |    // tensor_dim1[31:16]
                ((tile0 & 0xFFFFu) << 16));    // tile_dim0
  g1[4] = (int)(tile1 & 0xFFFFu);              // tile_dim1 (tile_dim2 = 0)
  g1[5] = (int)stride0;                        // tensor_dim0_stride[31:0]
  g1[6] = 0;                                   // stride0[47:32]=0, stride1 lo=0
  g1[7] = 0;

  i32x4 z4 = {};
#if __clang_major__ >= 23
  i32x8 z8 = {};
  __builtin_amdgcn_tensor_load_to_lds(g0, g1, z4, z4, z8, 0);
#else
  __builtin_amdgcn_tensor_load_to_lds(g0, g1, z4, z4, 0);
#endif
#else
  (void)gptr; (void)ldsOff; (void)tile0; (void)tile1;
  (void)tdim0; (void)tdim1; (void)stride0;
#endif
}

// ---------------------------------------------------------------------------
// Kernel 1: fused QKV projection.  q/k/v = x @ W^T, fp32 in -> bf16 out.
// 128x128 tile per workgroup, 8 waves each computing 32x64.
// (fp32->bf16 conversion during staging => cooperative loads, no TDM here)
// ---------------------------------------------------------------------------
__global__ __launch_bounds__(256) void qkv_gemm(
    const float* __restrict__ x,
    const float* __restrict__ Wq, const float* __restrict__ Wk,
    const float* __restrict__ Wv,
    __bf16* __restrict__ qb, __bf16* __restrict__ kb, __bf16* __restrict__ vb)
{
  const float* W = (blockIdx.z == 0) ? Wq : (blockIdx.z == 1) ? Wk : Wv;
  __bf16* dst    = (blockIdx.z == 0) ? qb : (blockIdx.z == 1) ? kb : vb;

  __shared__ __align__(32) __bf16 As[128 * 32];
  __shared__ __align__(32) __bf16 Bs[128 * 32];

  const int t    = threadIdx.x;
  const int lane = t & 31;
  const int wid  = t >> 5;        // 8 waves
  const int wm   = wid & 3;       // 4 waves along M (32 rows each)
  const int wn   = wid >> 2;      // 2 waves along N (64 cols each)
  const int half = lane >> 4;
  const int ln16 = lane & 15;
  const int m0   = blockIdx.y * 128;
  const int n0   = blockIdx.x * 128;

  v8f acc[2][4] = {};

  const int lrow = t >> 1;        // 0..127
  const int lcol = (t & 1) * 16;  // 0 or 16

  for (int k0 = 0; k0 < E_; k0 += 32) {
    __syncthreads();
    {
      const float* xs = x + (size_t)(m0 + lrow) * E_ + k0 + lcol;
      const float* ws = W + (size_t)(n0 + lrow) * E_ + k0 + lcol;
      if (k0 + 32 < E_) {
        __builtin_prefetch(xs + 32, 0, 1);
        __builtin_prefetch(ws + 32, 0, 1);
      }
#pragma unroll
      for (int j = 0; j < 16; ++j) {
        As[lrow * 32 + lcol + j] = (__bf16)xs[j];
        Bs[lrow * 32 + lcol + j] = (__bf16)ws[j];
      }
    }
    __syncthreads();

    v16bf afrag[2], bfrag[4];
#pragma unroll
    for (int mi = 0; mi < 2; ++mi)
      afrag[mi] = *(const v16bf*)&As[(wm * 32 + mi * 16 + ln16) * 32 + half * 16];
#pragma unroll
    for (int ni = 0; ni < 4; ++ni)
      bfrag[ni] = *(const v16bf*)&Bs[(wn * 64 + ni * 16 + ln16) * 32 + half * 16];

#pragma unroll
    for (int mi = 0; mi < 2; ++mi)
#pragma unroll
      for (int ni = 0; ni < 4; ++ni)
        acc[mi][ni] = wmma_bf16(afrag[mi], bfrag[ni], acc[mi][ni]);
  }

  // C layout: VGPR r holds row r (lanes 0-15) / row r+8 (lanes 16-31), col = lane&15.
#pragma unroll
  for (int mi = 0; mi < 2; ++mi)
    for (int ni = 0; ni < 4; ++ni)
#pragma unroll
      for (int r = 0; r < 8; ++r) {
        int row = m0 + wm * 32 + mi * 16 + r + half * 8;
        int col = n0 + wn * 64 + ni * 16 + ln16;
        dst[(size_t)row * E_ + col] = (__bf16)acc[mi][ni][r];
      }
}

// ---------------------------------------------------------------------------
// Kernel 2: RoPE on q and k (interleaved even/odd pairs, per reference).
// ---------------------------------------------------------------------------
__global__ __launch_bounds__(256) void rope_kernel(__bf16* __restrict__ qb,
                                                   __bf16* __restrict__ kb)
{
  const int nPairs = M_ * (E_ / 2);              // 2^21 per buffer
  int idx = blockIdx.x * blockDim.x + threadIdx.x;
  __bf16* buf = (idx < nPairs) ? qb : kb;
  int r   = idx & (nPairs - 1);
  int m   = r >> 9;                              // row 0..4095
  int p   = r & 511;                             // pair within row
  int ph  = p & 31;                              // pair within head (HD/2 = 32)
  int pos = m & (S_ - 1);

  const float lnb = 9.210340371976184f / 32.0f;  // ln(10000)/(HD/2)
  float inv = __expf(-lnb * (float)ph);
  float ang = (float)pos * inv;
  float sa, ca;
  __sincosf(ang, &sa, &ca);

  size_t o = (size_t)m * E_ + 2 * p;
  float x1 = (float)buf[o];
  float x2 = (float)buf[o + 1];
  buf[o]     = (__bf16)(x1 * ca - x2 * sa);
  buf[o + 1] = (__bf16)(x1 * sa + x2 * ca);
}

// ---------------------------------------------------------------------------
// Kernel 3: causal flash attention.  One workgroup = (b, h, 64 q rows),
// 4 waves x 16 q rows.  KV streamed in tiles of 32 with online softmax.
// K tile is DMA'd row-major by the TDM (wave 0 issues the descriptor and
// waits TENSORcnt while the other waves stage V transposed); V staged
// transposed so PV B-fragments are contiguous v16bf LDS reads.
// ---------------------------------------------------------------------------
__global__ __launch_bounds__(128) void flash_attn(
    const __bf16* __restrict__ qb, const __bf16* __restrict__ kb,
    const __bf16* __restrict__ vb, __bf16* __restrict__ ob)
{
  const int qt   = blockIdx.x;       // 0..31
  const int h    = blockIdx.y;       // 0..15
  const int b    = blockIdx.z;       // 0..1
  const int t    = threadIdx.x;
  const int lane = t & 31;
  const int wid  = t >> 5;           // 4 waves
  const int half = lane >> 4;
  const int ln16 = lane & 15;

  __shared__ __align__(32) __bf16 Kt[32 * 64];       // [kv][d]
  __shared__ __align__(32) __bf16 Vt[64 * 32];       // [d][kv] transposed
  __shared__ __align__(32) __bf16 Ps[4 * 16 * 32];   // per-wave P staging

  const int q0 = qt * 64;
  const int qw = q0 + wid * 16;
  const size_t rowbase = (size_t)b * S_;
  const float scale = 0.03125f;      // 1/sqrt(1024)

#ifdef USE_TDM
  const unsigned ktOff = (unsigned)(uintptr_t)(void*)Kt;  // LDS byte offset
#endif

  // Q fragments: 16x64 over d, two K=32 chunks
  v16bf aq[2];
#pragma unroll
  for (int kc = 0; kc < 2; ++kc)
    aq[kc] = *(const v16bf*)&qb[(rowbase + qw + ln16) * E_ + h * HD_ +
                                kc * 32 + half * 16];

  v8f acc[4] = {};
  float mrow[8], lrow[8];
#pragma unroll
  for (int r = 0; r < 8; ++r) { mrow[r] = -__builtin_inff(); lrow[r] = 0.0f; }

  const int ldrow = t >> 2;          // 0..31
  const int ldcol = (t & 3) * 16;    // 0,16,32,48

  for (int kv0 = 0; kv0 < q0 + 64; kv0 += 32) {
    __syncthreads();                 // prior tile fully consumed
#ifdef USE_TDM
    if (wid == 0)
      tdm_load_2d_bf16(&kb[(rowbase + kv0) * E_ + h * HD_], ktOff,
                       /*tile0=*/HD_, /*tile1=*/32,
                       /*tdim0=*/E_, /*tdim1=*/M_, /*stride0=*/E_);
#endif
    {
#ifndef USE_TDM
      const __bf16* ksrc = &kb[(rowbase + kv0 + ldrow) * E_ + h * HD_ + ldcol];
#endif
      const __bf16* vsrc = &vb[(rowbase + kv0 + ldrow) * E_ + h * HD_ + ldcol];
#pragma unroll
      for (int j = 0; j < 16; ++j) {
#ifndef USE_TDM
        Kt[ldrow * 64 + ldcol + j]   = ksrc[j];
#endif
        Vt[(ldcol + j) * 32 + ldrow] = vsrc[j];   // transpose into LDS
      }
    }
#ifdef USE_TDM
    if (wid == 0) __builtin_amdgcn_s_wait_tensorcnt(0);
#endif
    __syncthreads();

    // S = Q K^T : two 16x16 score tiles, K-dim = head dim (2 chunks of 32)
    v8f sc[2];
#pragma unroll
    for (int ct = 0; ct < 2; ++ct) {
      v8f z = {};
      sc[ct] = z;
#pragma unroll
      for (int kc = 0; kc < 2; ++kc) {
        v16bf bk = *(const v16bf*)&Kt[(ct * 16 + ln16) * 64 + kc * 32 + half * 16];
        sc[ct] = wmma_bf16(aq[kc], bk, sc[ct]);
      }
    }

    // scale + causal mask + online softmax update
    float pv[2][8];
#pragma unroll
    for (int r = 0; r < 8; ++r) {
      int qrow = qw + r + half * 8;
      float mx = -__builtin_inff();
#pragma unroll
      for (int ct = 0; ct < 2; ++ct) {
        int kcol = kv0 + ct * 16 + ln16;
        float s = (kcol <= qrow) ? sc[ct][r] * scale : -__builtin_inff();
        pv[ct][r] = s;
        mx = fmaxf(mx, s);
      }
#pragma unroll
      for (int d = 1; d < 16; d <<= 1) mx = fmaxf(mx, __shfl_xor(mx, d, 16));
      float mnew = fmaxf(mrow[r], mx);
      float corr = __expf(mrow[r] - mnew);     // exp(-inf)=0 on first tile
      float rsum = 0.0f;
#pragma unroll
      for (int ct = 0; ct < 2; ++ct) {
        float p = __expf(pv[ct][r] - mnew);
        pv[ct][r] = p;
        rsum += p;
      }
#pragma unroll
      for (int d = 1; d < 16; d <<= 1) rsum += __shfl_xor(rsum, d, 16);
      lrow[r] = lrow[r] * corr + rsum;
      mrow[r] = mnew;
#pragma unroll
      for (int nt = 0; nt < 4; ++nt) acc[nt][r] = acc[nt][r] * corr;
    }

    // reshape P: C layout -> A-fragment layout via LDS
#pragma unroll
    for (int ct = 0; ct < 2; ++ct)
#pragma unroll
      for (int r = 0; r < 8; ++r)
        Ps[wid * 512 + (r + half * 8) * 32 + ct * 16 + ln16] = (__bf16)pv[ct][r];
    __syncthreads();

    v16bf ap = *(const v16bf*)&Ps[wid * 512 + ln16 * 32 + half * 16];
#pragma unroll
    for (int nt = 0; nt < 4; ++nt) {
      v16bf bv = *(const v16bf*)&Vt[(nt * 16 + ln16) * 32 + half * 16];
      acc[nt] = wmma_bf16(ap, bv, acc[nt]);
    }
  }

  // normalize and store bf16 attention output
#pragma unroll
  for (int nt = 0; nt < 4; ++nt)
#pragma unroll
    for (int r = 0; r < 8; ++r) {
      int row = qw + r + half * 8;
      ob[(rowbase + row) * E_ + h * HD_ + nt * 16 + ln16] =
          (__bf16)(acc[nt][r] / lrow[r]);
    }
}

// ---------------------------------------------------------------------------
// Kernel 4: output projection.  out = attn @ Wo^T, bf16 A, fp32 result.
// A tile is bf16 row-major -> staged by TDM while all threads convert Wo.
// ---------------------------------------------------------------------------
__global__ __launch_bounds__(256) void out_gemm(
    const __bf16* __restrict__ A, const float* __restrict__ Wo,
    float* __restrict__ Out)
{
  __shared__ __align__(32) __bf16 As[128 * 32];
  __shared__ __align__(32) __bf16 Bs[128 * 32];

  const int t    = threadIdx.x;
  const int lane = t & 31;
  const int wid  = t >> 5;
  const int wm   = wid & 3;
  const int wn   = wid >> 2;
  const int half = lane >> 4;
  const int ln16 = lane & 15;
  const int m0   = blockIdx.y * 128;
  const int n0   = blockIdx.x * 128;

#ifdef USE_TDM
  const unsigned asOff = (unsigned)(uintptr_t)(void*)As;
#endif

  v8f acc[2][4] = {};

  const int lrow = t >> 1;
  const int lcol = (t & 1) * 16;

  for (int k0 = 0; k0 < E_; k0 += 32) {
    __syncthreads();
#ifdef USE_TDM
    if (wid == 0)
      tdm_load_2d_bf16(&A[(size_t)m0 * E_ + k0], asOff,
                       /*tile0=*/32, /*tile1=*/128,
                       /*tdim0=*/E_, /*tdim1=*/M_, /*stride0=*/E_);
#endif
    {
      const float* ws = Wo + (size_t)(n0 + lrow) * E_ + k0 + lcol;
      if (k0 + 32 < E_) __builtin_prefetch(ws + 32, 0, 1);
#pragma unroll
      for (int j = 0; j < 16; ++j)
        Bs[lrow * 32 + lcol + j] = (__bf16)ws[j];
#ifndef USE_TDM
      const __bf16* as = A + (size_t)(m0 + lrow) * E_ + k0 + lcol;
#pragma unroll
      for (int j = 0; j < 16; ++j)
        As[lrow * 32 + lcol + j] = as[j];
#endif
    }
#ifdef USE_TDM
    if (wid == 0) __builtin_amdgcn_s_wait_tensorcnt(0);
#endif
    __syncthreads();

    v16bf afrag[2], bfrag[4];
#pragma unroll
    for (int mi = 0; mi < 2; ++mi)
      afrag[mi] = *(const v16bf*)&As[(wm * 32 + mi * 16 + ln16) * 32 + half * 16];
#pragma unroll
    for (int ni = 0; ni < 4; ++ni)
      bfrag[ni] = *(const v16bf*)&Bs[(wn * 64 + ni * 16 + ln16) * 32 + half * 16];

#pragma unroll
    for (int mi = 0; mi < 2; ++mi)
#pragma unroll
      for (int ni = 0; ni < 4; ++ni)
        acc[mi][ni] = wmma_bf16(afrag[mi], bfrag[ni], acc[mi][ni]);
  }

#pragma unroll
  for (int mi = 0; mi < 2; ++mi)
    for (int ni = 0; ni < 4; ++ni)
#pragma unroll
      for (int r = 0; r < 8; ++r) {
        int row = m0 + wm * 32 + mi * 16 + r + half * 8;
        int col = n0 + wn * 64 + ni * 16 + ln16;
        Out[(size_t)row * E_ + col] = acc[mi][ni][r];
      }
}

// ---------------------------------------------------------------------------
extern "C" void kernel_launch(void* const* d_in, const int* in_sizes, int n_in,
                              void* d_out, int out_size, void* d_ws, size_t ws_size,
                              hipStream_t stream) {
  const float* x  = (const float*)d_in[0];
  const float* Wq = (const float*)d_in[1];
  const float* Wk = (const float*)d_in[2];
  const float* Wv = (const float*)d_in[3];
  const float* Wo = (const float*)d_in[4];
  float* out = (float*)d_out;

  // workspace: 4 bf16 buffers of M_*E_ (8 MB each -> 32 MB total)
  __bf16* qb = (__bf16*)d_ws;
  __bf16* kb = qb + (size_t)M_ * E_;
  __bf16* vb = kb + (size_t)M_ * E_;
  __bf16* ab = vb + (size_t)M_ * E_;

  dim3 g1(E_ / 128, M_ / 128, 3);
  qkv_gemm<<<g1, 256, 0, stream>>>(x, Wq, Wk, Wv, qb, kb, vb);

  int totalThreads = 2 * M_ * (E_ / 2);
  rope_kernel<<<totalThreads / 256, 256, 0, stream>>>(qb, kb);

  dim3 g3(S_ / 64, H_, B_);
  flash_attn<<<g3, 128, 0, stream>>>(qb, kb, vb, ab);

  dim3 g4(E_ / 128, M_ / 128);
  out_gemm<<<g4, 256, 0, stream>>>(ab, Wo, out);
}
